// CrossFLASH_46325517255394
// MI455X (gfx1250) — compile-verified, matrix operand order checked by reading, not run on hardware
//
#include <hip/hip_runtime.h>
#include <math.h>

typedef __attribute__((ext_vector_type(16))) _Float16 v16h;
typedef __attribute__((ext_vector_type(8)))  float    v8f;
typedef int v4i __attribute__((vector_size(16)));   // matches builtin's V4i pointee
typedef _Float16 h16;

#define AS1 __attribute__((address_space(1)))
#define AS3 __attribute__((address_space(3)))

// Probe the CDNA5 async global->LDS builtins (compile-safe fallback otherwise)
#if defined(__has_builtin)
#  if __has_builtin(__builtin_amdgcn_global_load_async_to_lds_b128)
#    define HAVE_ASYNC 1
#  endif
#  if __has_builtin(__builtin_amdgcn_s_wait_asynccnt)
#    define HAVE_WAITASYNC 1
#  endif
#endif
#ifndef HAVE_ASYNC
#  define HAVE_ASYNC 0
#endif
#ifndef HAVE_WAITASYNC
#  define HAVE_WAITASYNC 0
#endif

// Problem constants
#define BB   4
#define NN   4096
#define DIM  512
#define GRP  256
#define QK   128
#define HID  1024
#define MTOT (BB*NN)      // 16384

// GEMM tiling
#define BM 128
#define BN 64
#define BK 32
#define LDSTR 48          // padded LDS row stride in halfs: 96B, keeps b128 aligned

__device__ __forceinline__ float sigmoidf_(float x) { return 1.f / (1.f + __expf(-x)); }
__device__ __forceinline__ float siluf_(float x)    { return x * sigmoidf_(x); }

// Load one 16x32 f16 operand fragment (A or B^T layout, per CDNA5 ISA 7.12.2):
// lane L: row/col = L&15, K-halves koff = 8*(L>>4); elems 0..7 = K[koff..], 8..15 = K[16+koff..]
__device__ __forceinline__ v16h frag_load(const h16* p) {
  v16h f;
  float4* d = reinterpret_cast<float4*>(&f);
  d[0] = *reinterpret_cast<const float4*>(p);
  d[1] = *reinterpret_cast<const float4*>(p + 16);
  return f;
}

// C = A[M,K] * B[N,K]^T, batched over gridDim.z.
// A batch offset: z*sA.  B batch offset: (z/zdiv)*sB + (z%zdiv)*sB2.
// Double-buffered LDS pipeline: one barrier per K-step.
template<class Ep>
__global__ __launch_bounds__(256) void gemm_wmma(
    const h16* __restrict__ A, int lda, long long sA,
    const h16* __restrict__ B, int ldb, long long sB, long long sB2, int zdiv,
    int M, int N, int K, Ep ep)
{
  __shared__ __attribute__((aligned(16))) h16 As[2][BM * LDSTR];
  __shared__ __attribute__((aligned(16))) h16 Bs[2][BN * LDSTR];
  const int tid  = threadIdx.x;
  const int lane = tid & 31;
  const int wave = tid >> 5;
  const int wm   = wave >> 1;   // 0..3 -> 32-row strip
  const int wn   = wave & 1;    // 0..1 -> 32-col strip
  const int z    = blockIdx.z;

  const h16* Ab = A + (long long)z * sA + (long long)blockIdx.y * BM * lda;
  const h16* Bb = B + (long long)(z / zdiv) * sB + (long long)(z % zdiv) * sB2
                    + (long long)blockIdx.x * BN * ldb;

  const int arow = tid >> 1, aseg = tid & 1;   // A tile: 128 rows x (2 segs of 16 halfs)
  const int brow = tid >> 2, bseg = tid & 3;   // B tile:  64 rows x (4 segs of  8 halfs)

  v8f acc[2][2] = {};

  // stage one K-tile: global -> LDS (async-to-LDS when available)
  auto load_tile = [&](int buf, int k0) {
    const h16* asrc = Ab + (long long)arow * lda + k0 + aseg * 16;
    const h16* bsrc = Bb + (long long)brow * ldb + k0 + bseg * 8;
    h16* adst = &As[buf][arow * LDSTR + aseg * 16];
    h16* bdst = &Bs[buf][brow * LDSTR + bseg * 8];
#if HAVE_ASYNC
    __builtin_amdgcn_global_load_async_to_lds_b128(
        (AS1 v4i*)(h16*)asrc,       (AS3 v4i*)adst,       0, 0);
    __builtin_amdgcn_global_load_async_to_lds_b128(
        (AS1 v4i*)(h16*)(asrc + 8), (AS3 v4i*)(adst + 8), 0, 0);
    __builtin_amdgcn_global_load_async_to_lds_b128(
        (AS1 v4i*)(h16*)bsrc,       (AS3 v4i*)bdst,       0, 0);
#else
    const float4* a4 = reinterpret_cast<const float4*>(asrc);
    float4* ad = reinterpret_cast<float4*>(adst);
    ad[0] = a4[0]; ad[1] = a4[1];
    *reinterpret_cast<float4*>(bdst) = *reinterpret_cast<const float4*>(bsrc);
#endif
    __builtin_prefetch(asrc + BK, 0, 0);   // speculative: next-next K-tile
    __builtin_prefetch(bsrc + BK, 0, 0);
  };
  // make staged tile visible to the whole workgroup
  auto commit = [&]() {
#if HAVE_ASYNC
#  if HAVE_WAITASYNC
    __builtin_amdgcn_s_wait_asynccnt(0);
#  else
    asm volatile("s_wait_asynccnt 0x0" ::: "memory");
#  endif
#endif
    __syncthreads();
  };

  load_tile(0, 0);
  commit();
  int buf = 0;
  for (int k0 = 0; k0 < K; k0 += BK) {
    if (k0 + BK < K) load_tile(buf ^ 1, k0 + BK);

    const int r  = lane & 15;
    const int ko = (lane >> 4) * 8;
    v16h a0 = frag_load(&As[buf][(wm * 32 +  0 + r) * LDSTR + ko]);
    v16h a1 = frag_load(&As[buf][(wm * 32 + 16 + r) * LDSTR + ko]);
    v16h b0 = frag_load(&Bs[buf][(wn * 32 +  0 + r) * LDSTR + ko]);
    v16h b1 = frag_load(&Bs[buf][(wn * 32 + 16 + r) * LDSTR + ko]);

    acc[0][0] = __builtin_amdgcn_wmma_f32_16x16x32_f16(false, a0, false, b0, (short)0, acc[0][0], false, false);
    acc[0][1] = __builtin_amdgcn_wmma_f32_16x16x32_f16(false, a0, false, b1, (short)0, acc[0][1], false, false);
    acc[1][0] = __builtin_amdgcn_wmma_f32_16x16x32_f16(false, a1, false, b0, (short)0, acc[1][0], false, false);
    acc[1][1] = __builtin_amdgcn_wmma_f32_16x16x32_f16(false, a1, false, b1, (short)0, acc[1][1], false, false);

    commit();
    buf ^= 1;
  }

  const int mo = (lane >> 4) * 8, no = lane & 15;
  for (int i = 0; i < 2; ++i)
    for (int j = 0; j < 2; ++j)
      for (int rr = 0; rr < 8; ++rr) {
        int m = blockIdx.y * BM + wm * 32 + i * 16 + rr + mo;
        int n = blockIdx.x * BN + wn * 32 + j * 16 + no;
        if (m < M && n < N) ep(z, m, n, acc[i][j][rr]);
      }
}

// ---------------- epilogues ----------------
struct EpV {      // v = silu(.)  -> v_h [M,HID] and vT [B,HID,N]
  const float* bias; h16* vh; h16* vT;
  __device__ void operator()(int, int m, int n, float acc) const {
    float r = siluf_(acc + bias[n]);
    vh[(long long)m * HID + n] = (h16)r;
    vT[((long long)(m >> 12) * HID + n) * NN + (m & 4095)] = (h16)r;
  }
};
struct EpSiluRM { // cond = silu(.) row-major
  const float* bias; h16* out;
  __device__ void operator()(int, int m, int n, float acc) const {
    out[(long long)m * HID + n] = (h16)siluf_(acc + bias[n]);
  }
};
struct EpAddBias { // g1 = . + b
  const float* bias; h16* out;
  __device__ void operator()(int, int m, int n, float acc) const {
    out[(long long)m * HID + n] = (h16)(acc + bias[n]);
  }
};
struct EpGate {   // gate = g1 * sigmoid(. + b)
  const float* bias; const h16* g1; h16* gate;
  __device__ void operator()(int, int m, int n, float acc) const {
    long long i = (long long)m * HID + n;
    gate[i] = (h16)((float)g1[i] * sigmoidf_(acc + bias[n]));
  }
};
struct EpQ {      // q = silu(.); quad_q, lin_q row-major
  const float* bias; const float* qg; const float* qb; h16* qq; h16* lq;
  __device__ void operator()(int, int m, int n, float acc) const {
    float s = siluf_(acc + bias[n]);
    long long i = (long long)m * QK + n;
    qq[i] = (h16)(s * qg[n]      + qb[n]);
    lq[i] = (h16)(s * qg[QK + n] + qb[QK + n]);
  }
};
struct EpK {      // k = silu(.); quad_k row-major, lin_k transposed [B,QK,N]
  const float* bias; const float* kg; const float* kb; h16* qk; h16* lkT;
  __device__ void operator()(int, int m, int n, float acc) const {
    float s = siluf_(acc + bias[n]);
    qk[(long long)m * QK + n] = (h16)(s * kg[n] + kb[n]);
    lkT[((long long)(m >> 12) * QK + n) * NN + (m & 4095)] = (h16)(s * kg[QK + n] + kb[QK + n]);
  }
};
struct EpSim {    // attn = relu(sim/G + bias)^2
  const float* bias; h16* attn;
  __device__ void operator()(int z, int m, int n, float acc) const {
    float s = acc * (1.f / (float)GRP) + bias[m * GRP + n];
    float r = fmaxf(s, 0.f);
    attn[(long long)z * (GRP * GRP) + m * GRP + n] = (h16)(r * r);
  }
};
struct EpQuad {   // quad_out -> yacc f32
  float* yacc;
  __device__ void operator()(int z, int m, int n, float acc) const {
    int b = z >> 4, g = z & 15;
    yacc[((long long)b * NN + g * GRP + m) * HID + n] = acc;
  }
};
struct EpLinKV {  // lin_kv / N, stored transposed [B,HID,QK]
  h16* lkvT;
  __device__ void operator()(int z, int m, int n, float acc) const {
    lkvT[((long long)z * HID + n) * QK + m] = (h16)(acc * (1.f / (float)NN));
  }
};
struct EpLinOut { // y = gate * (quad + lin)  -> f16
  const float* yacc; const h16* gate; h16* yh;
  __device__ void operator()(int z, int m, int n, float acc) const {
    long long i = ((long long)z * NN + m) * HID + n;
    yh[i] = (h16)((yacc[i] + acc) * (float)gate[i]);
  }
};
struct EpOut {    // out = . + out_b + x (residual)
  const float* bias; const float* x; float* out;
  __device__ void operator()(int, int m, int n, float acc) const {
    long long i = (long long)m * DIM + n;
    out[i] = acc + bias[n] + x[i];
  }
};

// ---------------- helper kernels ----------------
__global__ __launch_bounds__(256) void ln_kernel(const float* __restrict__ in,
    const float* __restrict__ w, const float* __restrict__ b, h16* __restrict__ out) {
  __shared__ float ssum[256], ssq[256];
  const int t = threadIdx.x;
  const float* p = in + (long long)blockIdx.x * DIM;
  float a0 = p[t], a1 = p[t + 256];
  ssum[t] = a0 + a1;
  ssq[t]  = a0 * a0 + a1 * a1;
  __syncthreads();
  for (int s = 128; s > 0; s >>= 1) {
    if (t < s) { ssum[t] += ssum[t + s]; ssq[t] += ssq[t + s]; }
    __syncthreads();
  }
  float mean = ssum[0] * (1.f / (float)DIM);
  float var  = ssq[0] * (1.f / (float)DIM) - mean * mean;
  float rstd = rsqrtf(var + 1e-5f);
  h16* o = out + (long long)blockIdx.x * DIM;
  o[t]       = (h16)((a0 - mean) * rstd * w[t]       + b[t]);
  o[t + 256] = (h16)((a1 - mean) * rstd * w[t + 256] + b[t + 256]);
}

__global__ __launch_bounds__(256) void f2h_kernel(const float* __restrict__ src,
                                                  h16* __restrict__ dst, long long n) {
  long long i = (long long)blockIdx.x * 256 + threadIdx.x;
  if (i < n) dst[i] = (h16)src[i];
}

__global__ __launch_bounds__(256) void relbias_kernel(const float* __restrict__ rel_emb,
                                                      float* __restrict__ bias) {
  int i = blockIdx.x, j = threadIdx.x;     // [G,G]
  int npos = i - j;                        // -(k_pos - q_pos)
  int ret  = (npos < 0) ? 16 : 0;
  int na   = npos < 0 ? -npos : npos;
  int nac  = na < 1 ? 1 : na;
  int vl   = 8 + (int)(logf((float)nac * (1.f / 8.f)) / logf(16.f) * 8.f);
  if (vl > 15) vl = 15;
  int bucket = ret + ((na < 8) ? na : vl);
  bias[i * GRP + j] = rel_emb[bucket] * 11.313708498984761f;  // sqrt(128)
}

// ---------------- launch ----------------
extern "C" void kernel_launch(void* const* d_in, const int* in_sizes, int n_in,
                              void* d_out, int out_size, void* d_ws, size_t ws_size,
                              hipStream_t stream) {
  const float* x      = (const float*)d_in[0];
  const float* cnd    = (const float*)d_in[1];
  const float* ln_x_w = (const float*)d_in[2];
  const float* ln_x_b = (const float*)d_in[3];
  const float* ln_c_w = (const float*)d_in[4];
  const float* ln_c_b = (const float*)d_in[5];
  const float* xh_w   = (const float*)d_in[6];
  const float* xh_b   = (const float*)d_in[7];
  const float* ch_w   = (const float*)d_in[8];
  const float* ch_b   = (const float*)d_in[9];
  const float* q_w    = (const float*)d_in[10];
  const float* q_b    = (const float*)d_in[11];
  const float* k_w    = (const float*)d_in[12];
  const float* k_b    = (const float*)d_in[13];
  const float* glu1_w = (const float*)d_in[14];
  const float* glu1_b = (const float*)d_in[15];
  const float* glu2_w = (const float*)d_in[16];
  const float* glu2_b = (const float*)d_in[17];
  const float* qg     = (const float*)d_in[18];
  const float* qb     = (const float*)d_in[19];
  const float* kg     = (const float*)d_in[20];
  const float* kb     = (const float*)d_in[21];
  const float* out_w  = (const float*)d_in[22];
  const float* out_b  = (const float*)d_in[23];
  const float* rel_emb= (const float*)d_in[24];
  float* out = (float*)d_out;

  // bump allocator over d_ws (256B aligned)
  char* wp = (char*)d_ws;
  auto alloc = [&](long long bytes) -> char* {
    char* p = wp; wp += (bytes + 255) & ~255ll; return p;
  };
  h16* w_xh  = (h16*)alloc((long long)HID * DIM * 2);
  h16* w_ch  = (h16*)alloc((long long)HID * DIM * 2);
  h16* w_g1  = (h16*)alloc((long long)HID * HID * 2);
  h16* w_g2  = (h16*)alloc((long long)HID * HID * 2);
  h16* w_q   = (h16*)alloc((long long)QK * DIM * 2);
  h16* w_k   = (h16*)alloc((long long)QK * DIM * 2);
  h16* w_o   = (h16*)alloc((long long)DIM * HID * 2);
  h16* nx_h  = (h16*)alloc((long long)MTOT * DIM * 2);
  h16* nc_h  = (h16*)alloc((long long)MTOT * DIM * 2);
  h16* v_h   = (h16*)alloc((long long)MTOT * HID * 2);
  h16* vT_h  = (h16*)alloc((long long)MTOT * HID * 2);   // [B,HID,N]
  h16* c_h   = (h16*)alloc((long long)MTOT * HID * 2);
  h16* g1_h  = (h16*)alloc((long long)MTOT * HID * 2);
  h16* gate_h= (h16*)alloc((long long)MTOT * HID * 2);
  h16* y_h   = (h16*)alloc((long long)MTOT * HID * 2);
  h16* qq_h  = (h16*)alloc((long long)MTOT * QK * 2);
  h16* lq_h  = (h16*)alloc((long long)MTOT * QK * 2);
  h16* qk_h  = (h16*)alloc((long long)MTOT * QK * 2);
  h16* lkT_h = (h16*)alloc((long long)MTOT * QK * 2);    // [B,QK,N]
  h16* attn_h= (h16*)alloc((long long)BB * 16 * GRP * GRP * 2);
  h16* lkvT_h= (h16*)alloc((long long)BB * HID * QK * 2);// [B,HID,QK]
  float* bias_f = (float*)alloc((long long)GRP * GRP * 4);
  float* yacc_f = (float*)alloc((long long)MTOT * HID * 4);

  // 1) weight conversion f32 -> f16
  auto cvt = [&](const float* s, h16* d, long long n) {
    f2h_kernel<<<dim3((unsigned)((n + 255) / 256)), dim3(256), 0, stream>>>(s, d, n);
  };
  cvt(xh_w,   w_xh, (long long)HID * DIM);
  cvt(ch_w,   w_ch, (long long)HID * DIM);
  cvt(glu1_w, w_g1, (long long)HID * HID);
  cvt(glu2_w, w_g2, (long long)HID * HID);
  cvt(q_w,    w_q,  (long long)QK * DIM);
  cvt(k_w,    w_k,  (long long)QK * DIM);
  cvt(out_w,  w_o,  (long long)DIM * HID);

  // 2) LayerNorms -> f16
  ln_kernel<<<dim3(MTOT), dim3(256), 0, stream>>>(x,   ln_x_w, ln_x_b, nx_h);
  ln_kernel<<<dim3(MTOT), dim3(256), 0, stream>>>(cnd, ln_c_w, ln_c_b, nc_h);

  // 3) T5 relative position bias table
  relbias_kernel<<<dim3(GRP), dim3(256), 0, stream>>>(rel_emb, bias_f);

  const dim3 blk(256);
  // 4) v = silu(nx @ xh^T + b)  (also writes vT)
  gemm_wmma<<<dim3(HID / BN, MTOT / BM, 1), blk, 0, stream>>>(
      nx_h, DIM, 0ll, w_xh, DIM, 0ll, 0ll, 1, MTOT, HID, DIM, EpV{xh_b, v_h, vT_h});
  // 5) cond = silu(nc @ ch^T + b)
  gemm_wmma<<<dim3(HID / BN, MTOT / BM, 1), blk, 0, stream>>>(
      nc_h, DIM, 0ll, w_ch, DIM, 0ll, 0ll, 1, MTOT, HID, DIM, EpSiluRM{ch_b, c_h});
  // 6) g1 = v @ glu1^T + b
  gemm_wmma<<<dim3(HID / BN, MTOT / BM, 1), blk, 0, stream>>>(
      v_h, HID, 0ll, w_g1, HID, 0ll, 0ll, 1, MTOT, HID, HID, EpAddBias{glu1_b, g1_h});
  // 7) gate = g1 * sigmoid(cond @ glu2^T + b)
  gemm_wmma<<<dim3(HID / BN, MTOT / BM, 1), blk, 0, stream>>>(
      c_h, HID, 0ll, w_g2, HID, 0ll, 0ll, 1, MTOT, HID, HID, EpGate{glu2_b, g1_h, gate_h});
  // 8) q = silu(nc @ q^T + b) -> quad_q, lin_q
  gemm_wmma<<<dim3(QK / BN, MTOT / BM, 1), blk, 0, stream>>>(
      nc_h, DIM, 0ll, w_q, DIM, 0ll, 0ll, 1, MTOT, QK, DIM, EpQ{q_b, qg, qb, qq_h, lq_h});
  // 9) k = silu(nx @ k^T + b) -> quad_k, lin_k^T
  gemm_wmma<<<dim3(QK / BN, MTOT / BM, 1), blk, 0, stream>>>(
      nx_h, DIM, 0ll, w_k, DIM, 0ll, 0ll, 1, MTOT, QK, DIM, EpK{k_b, kg, kb, qk_h, lkT_h});
  // 10) attn = relu(qq @ qk^T / G + bias)^2, per (b,g): 64 batches
  gemm_wmma<<<dim3(GRP / BN, GRP / BM, BB * 16), blk, 0, stream>>>(
      qq_h, QK, (long long)GRP * QK, qk_h, QK, (long long)GRP * QK, 0ll, 1,
      GRP, GRP, QK, EpSim{bias_f, attn_h});
  // 11) quad_out = attn @ vg  (B operand: vT[b][:, g*G .. ])
  gemm_wmma<<<dim3(HID / BN, GRP / BM, BB * 16), blk, 0, stream>>>(
      attn_h, GRP, (long long)GRP * GRP, vT_h, NN, (long long)HID * NN, (long long)GRP, 16,
      GRP, HID, GRP, EpQuad{yacc_f});
  // 12) lin_kv = lk^T @ v / N  per batch -> stored transposed
  gemm_wmma<<<dim3(HID / BN, QK / BM, BB), blk, 0, stream>>>(
      lkT_h, NN, (long long)QK * NN, vT_h, NN, (long long)HID * NN, 0ll, 1,
      QK, HID, NN, EpLinKV{lkvT_h});
  // 13) y = gate * (quad_out + lq @ lin_kv)
  gemm_wmma<<<dim3(HID / BN, NN / BM, BB), blk, 0, stream>>>(
      lq_h, QK, (long long)NN * QK, lkvT_h, QK, (long long)HID * QK, 0ll, 1,
      NN, HID, QK, EpLinOut{yacc_f, gate_h, y_h});
  // 14) out = y @ out_w^T + out_b + x
  gemm_wmma<<<dim3(DIM / BN, MTOT / BM, 1), blk, 0, stream>>>(
      y_h, HID, 0ll, w_o, HID, 0ll, 0ll, 1, MTOT, DIM, HID, EpOut{out_b, x, out});

  (void)in_sizes; (void)n_in; (void)out_size; (void)ws_size;
}
